// FrequencyAttention_12841952215553
// MI455X (gfx1250) — matrix operand order repaired and checked by reading, not compiled, not run on hardware
//
#include <hip/hip_runtime.h>
#include <math.h>

typedef __attribute__((ext_vector_type(16))) _Float16 v16h;
typedef __attribute__((ext_vector_type(4)))  _Float16 v4h;
typedef __attribute__((ext_vector_type(8)))  float    v8f;
typedef __attribute__((ext_vector_type(4)))  unsigned int v4u;

union AFrag { v4u u[2]; v16h h; };

#define BDIM 16
#define TDIM 7
#define NDIM 2048
#define DDIM 128
#define MT   (BDIM*TDIM*NDIM)   /* 229376 rows */

__device__ __forceinline__ float gelu_exact(float x) {
    return 0.5f * x * (1.0f + erff(x * 0.70710678118654752f));
}

// Async copy 16B/lane from global straight into LDS (ASYNCcnt path).
__device__ __forceinline__ void async_ld_b128(unsigned int lds, const void* g) {
    asm volatile("global_load_async_to_lds_b128 %0, %1, off"
                 :: "v"(lds), "v"((unsigned long long)(uintptr_t)g)
                 : "memory");
}
__device__ __forceinline__ void wait_async0() {
    asm volatile("s_wait_asynccnt 0" ::: "memory");
}

// ---------------------------------------------------------------------------
// Pack concat [X | broadcast STE] into f16 A16[M,256] (one-time pass).
// ---------------------------------------------------------------------------
__global__ __launch_bounds__(256)
void pack_a16(const float* __restrict__ X, const float* __restrict__ STE,
              _Float16* __restrict__ A16)
{
    const int id = blockIdx.x * 256 + threadIdx.x;   // MT*64 threads, 4 chans each
    const int m  = id >> 6;
    const int g  = id & 63;
    const int c4 = g * 4;
    float4 f;
    if (c4 < 128) {
        f = ((const float4*)(X + (size_t)m * 128))[g];
    } else {
        const int bt = m >> 11;
        const int b  = bt / TDIM;
        const int n  = m & (NDIM - 1);
        f = ((const float4*)(STE + ((size_t)b * NDIM + n) * 128))[g - 32];
    }
    v4h h;
    h[0] = (_Float16)f.x; h[1] = (_Float16)f.y;
    h[2] = (_Float16)f.z; h[3] = (_Float16)f.w;
    *(v4h*)(A16 + (size_t)m * 256 + c4) = h;
}

// ---------------------------------------------------------------------------
// Pack W[K,128] fp32 into transposed f16 Wt[128,K].
// ---------------------------------------------------------------------------
__global__ __launch_bounds__(256)
void pack_wt(const float* __restrict__ W, _Float16* __restrict__ Wt, int K)
{
    const int id = blockIdx.x * 256 + threadIdx.x;   // K*128 threads
    const int c = id & 127;
    const int k = id >> 7;
    if (k < K) Wt[(size_t)c * K + k] = (_Float16)W[(size_t)k * 128 + c];
}

// ---------------------------------------------------------------------------
// C[M,128] = Ah[M,Kdim](f16) * Wt^T + bias, plus per-column sum/sumsq stats.
// 256 threads / 8 waves; 128x128 C tile; async double-buffered LDS staging.
// ---------------------------------------------------------------------------
__global__ __launch_bounds__(256)
void gemm_async_wmma(const _Float16* __restrict__ Ah,  // [M, Kdim]
                     const _Float16* __restrict__ Bt,  // [128, Kdim]  (= W^T)
                     const float* __restrict__ bias,
                     float* __restrict__ Y,
                     float* __restrict__ gS1, float* __restrict__ gS2,
                     int Kdim)
{
    __shared__ _Float16 As[2][128][32];   // 16KB
    __shared__ _Float16 Bs[2][128][32];   // 16KB
    __shared__ float sS1[128], sS2[128];

    const int tid    = threadIdx.x;
    const int wave   = tid >> 5;
    const int lane   = tid & 31;
    const int lane16 = lane & 15;
    const int hi     = lane >> 4;

    if (tid < 128) { sS1[tid] = 0.0f; sS2[tid] = 0.0f; }

    const int m0  = blockIdx.x * 128;
    const int row = tid >> 1;           // 0..127 (also the W^T column)
    const int seg = (tid & 1) * 16;     // halfs offset within the 32-wide k tile

    v8f acc[8];
    #pragma unroll
    for (int i = 0; i < 8; ++i) { v8f z = {}; acc[i] = z; }

    const int nk = Kdim >> 5;

    // issue one tile's async copies: 16B x 2 per thread for A, same for B
    auto issue = [&](int kt, int buf) {
        const int k0 = kt * 32;
        unsigned int la = (unsigned int)(uintptr_t)&As[buf][row][seg];
        const _Float16* ga = Ah + (size_t)(m0 + row) * Kdim + k0 + seg;
        async_ld_b128(la,      ga);
        async_ld_b128(la + 16, ga + 8);
        unsigned int lb = (unsigned int)(uintptr_t)&Bs[buf][row][seg];
        const _Float16* gb = Bt + (size_t)row * Kdim + k0 + seg;
        async_ld_b128(lb,      gb);
        async_ld_b128(lb + 16, gb + 8);
    };

    issue(0, 0);
    for (int kt = 0; kt < nk; ++kt) {
        wait_async0();
        __syncthreads();                         // tile[kt] fully in LDS
        if (kt + 1 < nk) issue(kt + 1, (kt + 1) & 1);   // prefetch next tile

        const int buf = kt & 1;
        AFrag af;
        const int arow = wave * 16 + lane16;
        af.u[0] = *(const v4u*)&As[buf][arow][hi * 8];
        af.u[1] = *(const v4u*)&As[buf][arow][hi * 8 + 16];

        AFrag bf[8];
        #pragma unroll
        for (int nt = 0; nt < 8; ++nt) {
            const int bcol = nt * 16 + lane16;
            bf[nt].u[0] = *(const v4u*)&Bs[buf][bcol][hi * 8];
            bf[nt].u[1] = *(const v4u*)&Bs[buf][bcol][hi * 8 + 16];
        }
        #pragma unroll
        for (int nt = 0; nt < 8; ++nt)
            acc[nt] = __builtin_amdgcn_wmma_f32_16x16x32_f16(
                false, af.h, false, bf[nt].h, (short)0, acc[nt], false, false);
    }

    // ---- bias + store + per-column stats ----
    const int rbase = m0 + wave * 16 + hi * 8;
    #pragma unroll
    for (int nt = 0; nt < 8; ++nt) {
        const int col = nt * 16 + lane16;
        const float bc = bias[col];
        float s1 = 0.0f, s2 = 0.0f;
        #pragma unroll
        for (int r = 0; r < 8; ++r) {
            float v = acc[nt][r] + bc;
            Y[(size_t)(rbase + r) * 128 + col] = v;
            s1 += v; s2 += v * v;
        }
        s1 += __shfl_xor(s1, 16, 32);
        s2 += __shfl_xor(s2, 16, 32);
        if (hi == 0) { atomicAdd(&sS1[col], s1); atomicAdd(&sS2[col], s2); }
    }
    __syncthreads();
    if (tid < 128) { atomicAdd(&gS1[tid], sS1[tid]); atomicAdd(&gS2[tid], sS2[tid]); }
}

// ---------------------------------------------------------------------------
__global__ void finalize_stats(const float* __restrict__ s1, const float* __restrict__ s2,
                               const float* __restrict__ g,  const float* __restrict__ bet,
                               float invM, float* __restrict__ scale, float* __restrict__ shift)
{
    int c = threadIdx.x;   // 128 threads
    float mu  = s1[c] * invM;
    float var = s2[c] * invM - mu * mu;
    float sc  = g[c] * rsqrtf(var + 1e-5f);
    scale[c] = sc;
    shift[c] = bet[c] - mu * sc;
}

// ---------------------------------------------------------------------------
// Attention: one thread per (b,n,h). BN+GELU on load; writes O in f16.
// ---------------------------------------------------------------------------
__global__ __launch_bounds__(256)
void attn_kernel(const float* __restrict__ Yq, const float* __restrict__ Yk,
                 const float* __restrict__ Yv,
                 const float* __restrict__ scq, const float* __restrict__ shq,
                 const float* __restrict__ sck, const float* __restrict__ shk,
                 const float* __restrict__ scv, const float* __restrict__ shv,
                 _Float16* __restrict__ O)
{
    const int id = blockIdx.x * blockDim.x + threadIdx.x;   // B*N*H = 524288
    const int h  = id & 15;
    const int n  = (id >> 4) & (NDIM - 1);
    const int b  = id >> 15;
    const int cb = h * 8;
    const size_t tstride = (size_t)NDIM * 128;
    const size_t base0   = ((size_t)(b * TDIM) * NDIM + n) * 128 + cb;

    float kk[TDIM][8], vv[TDIM][8];
    #pragma unroll
    for (int t = 0; t < TDIM; ++t) {
        #pragma unroll
        for (int c = 0; c < 8; ++c) {
            float xk = Yk[base0 + t * tstride + c];
            kk[t][c] = gelu_exact(xk * sck[cb + c] + shk[cb + c]);
            float xv = Yv[base0 + t * tstride + c];
            vv[t][c] = gelu_exact(xv * scv[cb + c] + shv[cb + c]);
        }
    }
    #pragma unroll
    for (int t = 0; t < TDIM; ++t) {
        float q[8];
        #pragma unroll
        for (int c = 0; c < 8; ++c) {
            float xq = Yq[base0 + t * tstride + c];
            q[c] = gelu_exact(xq * scq[cb + c] + shq[cb + c]);
        }
        float sc[TDIM]; float mx = -1e30f;
        #pragma unroll
        for (int s = 0; s < TDIM; ++s) {
            float d = 0.0f;
            #pragma unroll
            for (int c = 0; c < 8; ++c) d += q[c] * kk[s][c];
            sc[s] = d * 0.25f;                      // / sqrt(16)
            mx = fmaxf(mx, sc[s]);
        }
        float den = 0.0f;
        #pragma unroll
        for (int s = 0; s < TDIM; ++s) { sc[s] = __expf(sc[s] - mx); den += sc[s]; }
        const float inv = 1.0f / den;
        #pragma unroll
        for (int c = 0; c < 8; ++c) {
            float o = 0.0f;
            #pragma unroll
            for (int s = 0; s < TDIM; ++s) o += sc[s] * vv[s][c];
            O[base0 + t * tstride + c] = (_Float16)(o * inv);
        }
    }
}

// ---------------------------------------------------------------------------
// Final: BN+GELU on Yo fused with Linear(7->1) over time.
// ---------------------------------------------------------------------------
__global__ __launch_bounds__(256)
void timemix_kernel(const float* __restrict__ Yo,
                    const float* __restrict__ sco, const float* __restrict__ sho,
                    const float* __restrict__ lw,  const float* __restrict__ lb,
                    float* __restrict__ out)
{
    const int id = blockIdx.x * blockDim.x + threadIdx.x;   // B*N*D = 4194304
    const int d  = id & 127;
    const int n  = (id >> 7) & (NDIM - 1);
    const int b  = id >> 18;
    const float sc = sco[d], sh = sho[d];
    const size_t tstride = (size_t)NDIM * 128;
    const size_t base = ((size_t)(b * TDIM) * NDIM + n) * 128 + d;
    float acc = lb[0];
    #pragma unroll
    for (int t = 0; t < TDIM; ++t) {
        float x = Yo[base + t * tstride];
        acc += lw[t] * gelu_exact(x * sc + sh);
    }
    out[id] = acc;
}

// ---------------------------------------------------------------------------
extern "C" void kernel_launch(void* const* d_in, const int* in_sizes, int n_in,
                              void* d_out, int out_size, void* d_ws, size_t ws_size,
                              hipStream_t stream)
{
    const float* X    = (const float*)d_in[0];
    const float* STE  = (const float*)d_in[1];
    const float* wq   = (const float*)d_in[2];
    const float* bq   = (const float*)d_in[3];
    const float* gq   = (const float*)d_in[4];
    const float* betq = (const float*)d_in[5];
    const float* wk   = (const float*)d_in[6];
    const float* bk   = (const float*)d_in[7];
    const float* gk   = (const float*)d_in[8];
    const float* betk = (const float*)d_in[9];
    const float* wv   = (const float*)d_in[10];
    const float* bv   = (const float*)d_in[11];
    const float* gv   = (const float*)d_in[12];
    const float* betv = (const float*)d_in[13];
    const float* wo   = (const float*)d_in[14];
    const float* bo   = (const float*)d_in[15];
    const float* go   = (const float*)d_in[16];
    const float* beto = (const float*)d_in[17];
    const float* lw   = (const float*)d_in[18];
    const float* lb   = (const float*)d_in[19];

    char* base = (char*)d_ws;
    const size_t YBYTES = (size_t)MT * 128 * sizeof(float);     // 117,440,512
    float*    Yq  = (float*)(base);
    float*    Yk  = (float*)(base + YBYTES);
    float*    Yv  = (float*)(base + 2 * YBYTES);
    _Float16* A16 = (_Float16*)(base + 3 * YBYTES);             // M*256 f16 = YBYTES
    _Float16* O16 = (_Float16*)(base + 4 * YBYTES);             // M*128 f16 = YBYTES/2
    _Float16* wtq = (_Float16*)(base + 4 * YBYTES + YBYTES / 2);
    _Float16* wtk = wtq + 256 * 128;
    _Float16* wtv = wtk + 256 * 128;
    _Float16* wto = wtv + 256 * 128;
    float*    st  = (float*)(wto + 128 * 128);
    float*    Yo  = Yq;                          // alias: Yq dead after attention

    float *s1q = st,        *s2q = st + 128,
          *s1k = st + 256,  *s2k = st + 384,
          *s1v = st + 512,  *s2v = st + 640,
          *s1o = st + 768,  *s2o = st + 896;
    float *scq = st + 1024, *shq = st + 1152,
          *sck = st + 1280, *shk = st + 1408,
          *scv = st + 1536, *shv = st + 1664,
          *sco = st + 1792, *sho = st + 1920;

    hipMemsetAsync(st, 0, 1024 * sizeof(float), stream);

    // one-time f16 packing
    pack_a16<<<(MT * 64) / 256, 256, 0, stream>>>(X, STE, A16);
    pack_wt<<<(256 * 128) / 256, 256, 0, stream>>>(wq, wtq, 256);
    pack_wt<<<(256 * 128) / 256, 256, 0, stream>>>(wk, wtk, 256);
    pack_wt<<<(256 * 128) / 256, 256, 0, stream>>>(wv, wtv, 256);
    pack_wt<<<(128 * 128) / 256, 256, 0, stream>>>(wo, wto, 128);

    const int MBLK = MT / 128;                 // 1792
    const float invM = 1.0f / (float)MT;

    gemm_async_wmma<<<MBLK, 256, 0, stream>>>(A16, wtq, bq, Yq, s1q, s2q, 256);
    gemm_async_wmma<<<MBLK, 256, 0, stream>>>(A16, wtk, bk, Yk, s1k, s2k, 256);
    gemm_async_wmma<<<MBLK, 256, 0, stream>>>(A16, wtv, bv, Yv, s1v, s2v, 256);

    finalize_stats<<<1, 128, 0, stream>>>(s1q, s2q, gq, betq, invM, scq, shq);
    finalize_stats<<<1, 128, 0, stream>>>(s1k, s2k, gk, betk, invM, sck, shk);
    finalize_stats<<<1, 128, 0, stream>>>(s1v, s2v, gv, betv, invM, scv, shv);

    attn_kernel<<<(BDIM * NDIM * 16) / 256, 256, 0, stream>>>(
        Yq, Yk, Yv, scq, shq, sck, shk, scv, shv, O16);

    gemm_async_wmma<<<MBLK, 256, 0, stream>>>(O16, wto, bo, Yo, s1o, s2o, 128);
    finalize_stats<<<1, 128, 0, stream>>>(s1o, s2o, go, beto, invM, sco, sho);

    timemix_kernel<<<(BDIM * NDIM * DDIM) / 256, 256, 0, stream>>>(
        Yo, sco, sho, lw, lb, (float*)d_out);
}
